// GRUNet_2061584302683
// MI455X (gfx1250) — matrix operand order repaired and checked by reading, not compiled, not run on hardware
//
#include <hip/hip_runtime.h>
#include <hip/hip_bf16.h>

typedef __attribute__((ext_vector_type(16))) _Float16 v16h;
typedef __attribute__((ext_vector_type(8)))  _Float16 v8h;
typedef __attribute__((ext_vector_type(8)))  float    v8f;

// ---------------------------------------------------------------------------
// WMMA fragment helpers (CDNA5 wave32 layouts, CDNA5 ISA §7.12.2)
// ---------------------------------------------------------------------------
__device__ __forceinline__ int kmapA(int v, int hi) {
    return (v < 4 ? 2 * v : 16 + 2 * (v - 4)) + hi * 8;
}

__device__ __forceinline__ v16h load_a_f32(const float* A, int lda, int m0, int k0) {
    const int l = threadIdx.x & 31;
    const int m = m0 + (l & 15);
    const int hi = l >> 4;
    v16h a;
#pragma unroll
    for (int v = 0; v < 8; ++v) {
        const int k = k0 + kmapA(v, hi);
        a[2 * v]     = (_Float16)A[(long)m * lda + k];
        a[2 * v + 1] = (_Float16)A[(long)m * lda + k + 1];
    }
    return a;
}

__device__ __forceinline__ v16h load_a_f16(const _Float16* A, long lda, int m0, long k0) {
    const int l = threadIdx.x & 31;
    const int m = m0 + (l & 15);
    const int hi = l >> 4;
    v16h a;
#pragma unroll
    for (int v = 0; v < 8; ++v) {
        const long k = k0 + kmapA(v, hi);
        a[2 * v]     = A[(long)m * lda + k];
        a[2 * v + 1] = A[(long)m * lda + k + 1];
    }
    return a;
}

// B fragment from weight matrix W[N,K] row-major (computing A @ W^T)
__device__ __forceinline__ v16h load_b_w(const float* W, int K, int n0, int k0) {
    const int l = threadIdx.x & 31;
    const int n = n0 + (l & 15);
    const int hi = l >> 4;
    v16h b;
#pragma unroll
    for (int v = 0; v < 8; ++v) {
        const int k = k0 + kmapA(v, hi);
        b[2 * v]     = (_Float16)W[(long)n * K + k];
        b[2 * v + 1] = (_Float16)W[(long)n * K + k + 1];
    }
    return b;
}

__device__ __forceinline__ v8f wmma_f16(v16h a, v16h b, v8f c) {
    return __builtin_amdgcn_wmma_f32_16x16x32_f16(false, a, false, b, (short)0, c,
                                                  false, false);
}

// Fast bounded-input transcendentals: v_exp_f32 + v_rcp_f32, no IEEE div fixup.
__device__ __forceinline__ float fast_sigmoid(float x) {
    return __builtin_amdgcn_rcpf(1.f + __expf(-x));
}
__device__ __forceinline__ float fast_tanh(float x) {
    return 1.f - 2.f * __builtin_amdgcn_rcpf(__expf(2.f * x) + 1.f);
}

// ---------------------------------------------------------------------------
// Generic batched GEMM with bias: C[M,N] = A[M,K] @ W[N,K]^T + bias
// ---------------------------------------------------------------------------
template <int K, bool A_F16, bool C_F16>
__global__ __launch_bounds__(256) void wmma_gemm_bias(
    const float* __restrict__ Af32, const _Float16* __restrict__ Af16,
    const float* __restrict__ W, const float* __restrict__ bias,
    float* __restrict__ Cf32, _Float16* __restrict__ Cf16, int M, int N)
{
    const int wave = threadIdx.x >> 5;
    const int tile = blockIdx.x * (blockDim.x >> 5) + wave;
    const int tiles_n = N >> 4;
    const int tm = tile / tiles_n;
    const int tn = tile - tm * tiles_n;
    const int m0 = tm << 4, n0 = tn << 4;

    v8f acc = {};
#pragma unroll
    for (int k0 = 0; k0 < K; k0 += 32) {
        v16h a = A_F16 ? load_a_f16(Af16, K, m0, k0) : load_a_f32(Af32, K, m0, k0);
        v16h b = load_b_w(W, K, n0, k0);
        acc = wmma_f16(a, b, acc);
    }

    const int l = threadIdx.x & 31;
    const int n = n0 + (l & 15);
    const int hi = l >> 4;
    const float bv = bias[n];
#pragma unroll
    for (int j = 0; j < 8; ++j) {
        const int m = m0 + 8 * hi + j;
        const float r = acc[j] + bv;
        if (C_F16) Cf16[(long)m * N + n] = (_Float16)r;
        else       Cf32[(long)m * N + n] = r;
    }
}

__global__ void init_flags(int* f) { f[threadIdx.x] = 0; }

// ---------------------------------------------------------------------------
// Pipelined 2-layer GRU scan. 8 blocks x 384 threads (12 waves).
//   blocks 0..3 : layer 0, batch group bg = blockIdx.x      (producer)
//   blocks 4..7 : layer 1, batch group bg = blockIdx.x - 4  (consumer)
// Each wave owns 2 N-tiles of the 384-wide gate projection. Weights live in
// registers as WMMA B-fragments (layer 1: both W_hh1 and W_ih1 -> the input
// projection is fused into the scan; no separate gi1 pass). h lives in LDS
// (f16) for the WMMA A-path; each gate thread keeps its h slice in registers.
// Layer-1 consumers overlap their h-part WMMAs with the wait for y0[t], which
// arrives through L2 (192 MB: guaranteed resident) via release/acquire flags.
// ---------------------------------------------------------------------------
__global__ __launch_bounds__(384) void gru_pipeline(
    const _Float16* __restrict__ gi0,  // [B,T,384] precomputed x@W_ih0^T + b_ih0
    const float* __restrict__ Whh0, const float* __restrict__ bhh0,
    const float* __restrict__ Wih1, const float* __restrict__ bih1,
    const float* __restrict__ Whh1, const float* __restrict__ bhh1,
    _Float16* __restrict__ y0,         // [B,T,128] layer-0 output (f16)
    _Float16* __restrict__ y1,         // [B,T,128] layer-1 output (f16)
    float* __restrict__ hfin,          // [2,B,128] slice of d_out
    int* __restrict__ flags,           // 4 groups x 32-int stride
    int T)
{
    __shared__ __align__(32) _Float16 h_lds[16 * 128];
    __shared__ __align__(32) float gh_lds[16 * 384];
    __shared__ __align__(32) float gi_lds[16 * 384];   // layer-1 only

    const int layer = blockIdx.x >> 2;
    const int bg    = blockIdx.x & 3;
    const int tid   = threadIdx.x;
    const int wave  = tid >> 5;          // 0..11
    const int lane  = tid & 31;
    const int B     = 64;

    // Recurrent-weight B fragments: 2 N-tiles x 4 K-chunks per wave.
    const float* Whh = layer ? Whh1 : Whh0;
    v16h bh[2][4];
#pragma unroll
    for (int nt = 0; nt < 2; ++nt)
#pragma unroll
        for (int kc = 0; kc < 4; ++kc)
            bh[nt][kc] = load_b_w(Whh, 128, (wave * 2 + nt) * 16, kc * 32);

    // Layer-1: input-projection B fragments too (fused gi1).
    v16h bi[2][4];
    if (layer) {
#pragma unroll
        for (int nt = 0; nt < 2; ++nt)
#pragma unroll
            for (int kc = 0; kc < 4; ++kc)
                bi[nt][kc] = load_b_w(Wih1, 128, (wave * 2 + nt) * 16, kc * 32);
    }

    for (int i = tid; i < 16 * 128; i += 384) h_lds[i] = (_Float16)0.f;

    // Gate-math threads: tid<256 -> (row m_g, 8 channels at c0)
    const bool gth  = tid < 256;
    const int m_g   = (tid & 255) >> 4;
    const int c0    = (tid & 15) * 8;
    const int b_idx = bg * 16 + m_g;
    const float* bhh = layer ? bhh1 : bhh0;
    const v8f bhr = *(const v8f*)&bhh[c0];
    const v8f bhz = *(const v8f*)&bhh[128 + c0];
    const v8f bhn = *(const v8f*)&bhh[256 + c0];
    v8f bir = {}, biz = {}, bin = {};
    if (layer) {
        bir = *(const v8f*)&bih1[c0];
        biz = *(const v8f*)&bih1[128 + c0];
        bin = *(const v8f*)&bih1[256 + c0];
    }

    v8h hcur = {};
    const float* ghrow = &gh_lds[m_g * 384];
    const float* girow = &gi_lds[m_g * 384];
    int* flag = flags + bg * 32;

    __syncthreads();

    if (layer == 0) {
        // ============================ LAYER 0 (producer) ====================
        for (int t = 0; t < T; ++t) {
            v8h gir, giz, gin;
            if (gth) {  // issue gi loads early; HBM latency hides behind WMMAs
                const long gib = ((long)b_idx * T + t) * 384;
                gir = *(const v8h*)&gi0[gib + c0];
                giz = *(const v8h*)&gi0[gib + 128 + c0];
                gin = *(const v8h*)&gi0[gib + 256 + c0];
                __builtin_prefetch((const void*)(gi0 + gib + 8L * 384 + c0), 0, 3);
                __builtin_prefetch((const void*)(gi0 + gib + 8L * 384 + 128 + c0), 0, 3);
                __builtin_prefetch((const void*)(gi0 + gib + 8L * 384 + 256 + c0), 0, 3);
            }
            // A fragments of h from LDS
            v16h ah[4];
#pragma unroll
            for (int kc = 0; kc < 4; ++kc) {
                const int m = lane & 15, hi = lane >> 4;
                v16h a;
#pragma unroll
                for (int v = 0; v < 8; ++v) {
                    const int k = kc * 32 + kmapA(v, hi);
                    a[2 * v]     = h_lds[m * 128 + k];
                    a[2 * v + 1] = h_lds[m * 128 + k + 1];
                }
                ah[kc] = a;
            }
            // gh = h @ W_hh0^T : 8 WMMAs/wave
#pragma unroll
            for (int nt = 0; nt < 2; ++nt) {
                v8f acc = {};
#pragma unroll
                for (int kc = 0; kc < 4; ++kc) acc = wmma_f16(ah[kc], bh[nt][kc], acc);
                const int n  = (wave * 2 + nt) * 16 + (lane & 15);
                const int hi = lane >> 4;
#pragma unroll
                for (int j = 0; j < 8; ++j) gh_lds[(8 * hi + j) * 384 + n] = acc[j];
            }
            __syncthreads();

            if (gth) {
                const v8f ghr = *(const v8f*)&ghrow[c0];
                const v8f ghz = *(const v8f*)&ghrow[128 + c0];
                const v8f ghn = *(const v8f*)&ghrow[256 + c0];
                v8h hnew;
#pragma unroll
                for (int i = 0; i < 8; ++i) {
                    const float r  = fast_sigmoid((float)gir[i] + ghr[i] + bhr[i]);
                    const float z  = fast_sigmoid((float)giz[i] + ghz[i] + bhz[i]);
                    const float nn = fast_tanh((float)gin[i] + r * (ghn[i] + bhn[i]));
                    hnew[i] = (_Float16)((1.f - z) * nn + z * (float)hcur[i]);
                }
                hcur = hnew;
                *(v8h*)&h_lds[m_g * 128 + c0] = hnew;
                *(v8h*)&y0[((long)b_idx * T + t) * 128 + c0] = hnew;
            }
            __threadfence();    // y0 tile visible device-wide before signaling
            __syncthreads();
            if (tid == 0)
                __hip_atomic_store(flag, t + 1, __ATOMIC_RELEASE,
                                   __HIP_MEMORY_SCOPE_AGENT);
        }
        if (gth)
#pragma unroll
            for (int i = 0; i < 8; ++i)
                hfin[(long)b_idx * 128 + c0 + i] = (float)hcur[i];
    } else {
        // ============================ LAYER 1 (consumer) ====================
        for (int t = 0; t < T; ++t) {
            // h-part first: overlaps with waiting for the producer's y0[t]
            v16h ah[4];
#pragma unroll
            for (int kc = 0; kc < 4; ++kc) {
                const int m = lane & 15, hi = lane >> 4;
                v16h a;
#pragma unroll
                for (int v = 0; v < 8; ++v) {
                    const int k = kc * 32 + kmapA(v, hi);
                    a[2 * v]     = h_lds[m * 128 + k];
                    a[2 * v + 1] = h_lds[m * 128 + k + 1];
                }
                ah[kc] = a;
            }
#pragma unroll
            for (int nt = 0; nt < 2; ++nt) {
                v8f acc = {};
#pragma unroll
                for (int kc = 0; kc < 4; ++kc) acc = wmma_f16(ah[kc], bh[nt][kc], acc);
                const int n  = (wave * 2 + nt) * 16 + (lane & 15);
                const int hi = lane >> 4;
#pragma unroll
                for (int j = 0; j < 8; ++j) gh_lds[(8 * hi + j) * 384 + n] = acc[j];
            }
            // acquire y0[t] from producer (L2-resident)
            if (tid == 0) {
                while (__hip_atomic_load(flag, __ATOMIC_ACQUIRE,
                                         __HIP_MEMORY_SCOPE_AGENT) <= t)
                    __builtin_amdgcn_s_sleep(2);
            }
            __syncthreads();

            // i-part: gi1 = y0_t @ W_ih1^T, A-fragments straight from global
            v16h ay[4];
#pragma unroll
            for (int kc = 0; kc < 4; ++kc)
                ay[kc] = load_a_f16(y0, (long)T * 128, bg * 16,
                                    (long)t * 128 + kc * 32);
#pragma unroll
            for (int nt = 0; nt < 2; ++nt) {
                v8f acc = {};
#pragma unroll
                for (int kc = 0; kc < 4; ++kc) acc = wmma_f16(ay[kc], bi[nt][kc], acc);
                const int n  = (wave * 2 + nt) * 16 + (lane & 15);
                const int hi = lane >> 4;
#pragma unroll
                for (int j = 0; j < 8; ++j) gi_lds[(8 * hi + j) * 384 + n] = acc[j];
            }
            __syncthreads();

            if (gth) {
                const v8f ghr = *(const v8f*)&ghrow[c0];
                const v8f ghz = *(const v8f*)&ghrow[128 + c0];
                const v8f ghn = *(const v8f*)&ghrow[256 + c0];
                const v8f qir = *(const v8f*)&girow[c0];
                const v8f qiz = *(const v8f*)&girow[128 + c0];
                const v8f qin = *(const v8f*)&girow[256 + c0];
                v8h hnew;
#pragma unroll
                for (int i = 0; i < 8; ++i) {
                    const float r  = fast_sigmoid(qir[i] + bir[i] + ghr[i] + bhr[i]);
                    const float z  = fast_sigmoid(qiz[i] + biz[i] + ghz[i] + bhz[i]);
                    const float nn = fast_tanh(qin[i] + bin[i] + r * (ghn[i] + bhn[i]));
                    hnew[i] = (_Float16)((1.f - z) * nn + z * (float)hcur[i]);
                }
                hcur = hnew;
                *(v8h*)&h_lds[m_g * 128 + c0] = hnew;
                *(v8h*)&y1[((long)b_idx * T + t) * 128 + c0] = hnew;
            }
            __syncthreads();
        }
        if (gth)
#pragma unroll
            for (int i = 0; i < 8; ++i)
                hfin[(long)(B + b_idx) * 128 + c0 + i] = (float)hcur[i];
    }
}

// ---------------------------------------------------------------------------
extern "C" void kernel_launch(void* const* d_in, const int* in_sizes, int n_in,
                              void* d_out, int out_size, void* d_ws, size_t ws_size,
                              hipStream_t stream) {
    const float* x     = (const float*)d_in[0];
    const float* W_ih0 = (const float*)d_in[1];
    const float* W_hh0 = (const float*)d_in[2];
    const float* b_ih0 = (const float*)d_in[3];
    const float* b_hh0 = (const float*)d_in[4];
    const float* W_ih1 = (const float*)d_in[5];
    const float* W_hh1 = (const float*)d_in[6];
    const float* b_ih1 = (const float*)d_in[7];
    const float* b_hh1 = (const float*)d_in[8];
    const float* fc_w  = (const float*)d_in[9];
    const float* fc_b  = (const float*)d_in[10];

    const int B = 64, T = 4096, OUT = 32;
    const long M = (long)B * T;  // 262144

    // Workspace: gi0, y0, y1, flags
    char* ws = (char*)d_ws;
    _Float16* gi0 = (_Float16*)ws;                                    // [M,384]
    _Float16* y0  = (_Float16*)(ws + (size_t)M * 384 * 2);            // [M,128]
    _Float16* y1  = (_Float16*)(ws + (size_t)M * 384 * 2 + (size_t)M * 128 * 2);
    int* flags    = (int*)(ws + (size_t)M * 384 * 2 + 2 * (size_t)M * 128 * 2);

    float* out  = (float*)d_out;       // [B,T,OUT]
    float* hfin = out + M * OUT;       // [2,B,128]

    const int tiles384 = (int)(M / 16) * (384 / 16);  // 393216
    const int tiles32  = (int)(M / 16) * (OUT / 16);  // 32768

    // reset producer/consumer flags (deterministic across graph replays)
    init_flags<<<1, 128, 0, stream>>>(flags);

    // K1: gi0 = x @ W_ih0^T + b_ih0   (K=32, f32 A -> f16 C)
    wmma_gemm_bias<32, false, true><<<tiles384 / 8, 256, 0, stream>>>(
        x, nullptr, W_ih0, b_ih0, nullptr, gi0, (int)M, 384);

    // P: both GRU layers, software-pipelined through L2 (layer-1 input
    //    projection fused into the scan; no bulk gi1 pass)
    gru_pipeline<<<8, 384, 0, stream>>>(gi0, W_hh0, b_hh0, W_ih1, b_ih1,
                                        W_hh1, b_hh1, y0, y1, hfin, flags, T);

    // K5: out = y1 @ fc_w^T + fc_b    (K=128, f16 A -> f32 C into d_out)
    wmma_gemm_bias<128, true, false><<<tiles32 / 8, 256, 0, stream>>>(
        nullptr, y1, fc_w, fc_b, out, nullptr, (int)M, OUT);
}
